// LSTM_2052994368001
// MI455X (gfx1250) — compile-verified, matrix operand order checked by reading, not run on hardware
//
#include <hip/hip_runtime.h>
#include <math.h>

// ---------------- CDNA5 WMMA types ----------------
typedef __attribute__((ext_vector_type(16))) _Float16 v16h;
typedef __attribute__((ext_vector_type(8)))  _Float16 v8h;
typedef __attribute__((ext_vector_type(8)))  float    v8f;

// Problem constants (from reference)
static constexpr int BB = 64;      // batch
static constexpr int TT = 512;     // time
static constexpr int DD = 512;     // input dim
static constexpr int HH = 512;     // hidden
static constexpr int GG = 4 * HH;  // 2048 gates
static constexpr int MROWS = BB * TT;       // 32768 rows of x
static constexpr int NBLK_SCAN = (BB / 16) * (HH / 16);  // 128 persistent blocks

// ---------------------------------------------------------------
// Tile loader: 16x32 f16 operand for v_wmma_f32_16x16x32_f16.
// ISA layout (CDNA5 ISA, 16-bit A 16x32):
//   lanes 0-15 : M=lane,   halves K in [0..7]  (V0..3) and [16..23] (V4..7)
//   lanes 16-31: M=lane-16, halves K in [8..15] and [24..31]
// => per lane: two contiguous 8-half (16B) chunks at K offsets hi*8, hi*8+16.
// Same function loads B tiles from a pre-transposed W^T[N,K] (lane = N).
// ---------------------------------------------------------------
__device__ __forceinline__ v16h pack16(v8h c0, v8h c1) {
  v16h r;
#pragma unroll
  for (int i = 0; i < 8; ++i) { r[i] = c0[i]; r[i + 8] = c1[i]; }
  return r;
}

__device__ __forceinline__ v16h load_tile_16x32(const _Float16* __restrict__ base,
                                                int ld, int lane) {
  const int row = lane & 15;
  const int hi  = lane >> 4;            // 0 or 1
  const _Float16* p = base + (size_t)row * ld + hi * 8;
  return pack16(*reinterpret_cast<const v8h*>(p),
                *reinterpret_cast<const v8h*>(p + 16));
}

__device__ __forceinline__ float sigmoidf_dev(float x) {
  return 1.0f / (1.0f + __expf(-x));
}

// ---------------- conversion / init kernels ----------------
__global__ void cvt_f32_to_f16(const float* __restrict__ in,
                               _Float16* __restrict__ out, int n) {
  int i = blockIdx.x * blockDim.x + threadIdx.x;
  if (i < n) out[i] = (_Float16)in[i];
}

// W [K, N] f32 row-major -> WT [N, K] f16 row-major
__global__ void transpose_w_f16(const float* __restrict__ w,
                                _Float16* __restrict__ wt, int K, int N) {
  int i = blockIdx.x * blockDim.x + threadIdx.x;   // i = n*K + k
  if (i < K * N) {
    int k = i & (K - 1);
    int n = i / K;
    wt[i] = (_Float16)w[(size_t)k * N + n];
  }
}

__global__ void init_state(_Float16* __restrict__ hA, _Float16* __restrict__ hB,
                           float* __restrict__ c, unsigned* __restrict__ bar, int n) {
  int i = blockIdx.x * blockDim.x + threadIdx.x;
  if (i < n) { hA[i] = (_Float16)0.0f; hB[i] = (_Float16)0.0f; c[i] = 0.0f; }
  if (i == 0) *bar = 0u;
}

// ---------------------------------------------------------------
// Phase 1: x_proj[t,b,g] = sum_k x16[b*T+t, k] * WihT[g, k] + bias[g]
// Register-blocked 2(M)x4(N) tiles per wave: per K-step, 12 b128 loads
// feed 8 WMMAs (load:WMMA = 1.5), A reused 4x, B reused 2x.
// Block = 128 threads = 4 waves; grid = (MROWS/32, 8).
// ---------------------------------------------------------------
__global__ __launch_bounds__(128)
void gemm_xproj(const _Float16* __restrict__ x16,      // [32768, 512]
                const _Float16* __restrict__ wih_t,    // [2048, 512]
                const float*    __restrict__ bias,     // [2048]
                float* __restrict__ xproj) {           // [T, B, 4H]
  const int lane = threadIdx.x & 31;
  const int wave = threadIdx.x >> 5;
  const int m0 = blockIdx.x * 32;                      // two 16-row M tiles
  const int ngroup = blockIdx.y * 4 + wave;            // 0..31
  const int n0 = ngroup * 64;                          // four 16-col N tiles

  v8f acc[2][4];
#pragma unroll
  for (int mt = 0; mt < 2; ++mt)
#pragma unroll
    for (int nt = 0; nt < 4; ++nt) acc[mt][nt] = (v8f)(0.0f);

  const _Float16* abase0 = x16 + (size_t)m0 * DD;
  const _Float16* abase1 = x16 + (size_t)(m0 + 16) * DD;
  const _Float16* bbase[4];
#pragma unroll
  for (int nt = 0; nt < 4; ++nt) bbase[nt] = wih_t + (size_t)(n0 + nt * 16) * DD;

#pragma unroll
  for (int kk = 0; kk < 16; ++kk) {
    const int k = kk * 32;
    v16h a0 = load_tile_16x32(abase0 + k, DD, lane);
    v16h a1 = load_tile_16x32(abase1 + k, DD, lane);
    v16h b[4];
#pragma unroll
    for (int nt = 0; nt < 4; ++nt) b[nt] = load_tile_16x32(bbase[nt] + k, DD, lane);
#pragma unroll
    for (int nt = 0; nt < 4; ++nt) {
      acc[0][nt] = __builtin_amdgcn_wmma_f32_16x16x32_f16(
          false, a0, false, b[nt], (short)0, acc[0][nt], false, false);
      acc[1][nt] = __builtin_amdgcn_wmma_f32_16x16x32_f16(
          false, a1, false, b[nt], (short)0, acc[1][nt], false, false);
    }
  }

  const int rlo = (lane >> 4) * 8;
  const int cl  = lane & 15;
#pragma unroll
  for (int mt = 0; mt < 2; ++mt) {
#pragma unroll
    for (int nt = 0; nt < 4; ++nt) {
      int col = n0 + nt * 16 + cl;
      float bv = bias[col];
#pragma unroll
      for (int j = 0; j < 8; ++j) {
        int gr = m0 + mt * 16 + j + rlo;  // flat row in [B*T)
        int b  = gr >> 9;                 // / T
        int t  = gr & (TT - 1);           // % T
        xproj[(size_t)t * (BB * GG) + (size_t)b * GG + col] = acc[mt][nt][j] + bv;
      }
    }
  }
}

// ---------------------------------------------------------------
// Phase 2: persistent cooperative scan. 128 single-wave workgroups.
// Each wave owns a 16(batch) x 16(hidden) tile, stages its four gate
// B-tiles (W_hh^T slice, 64KB) into LDS ONCE, then iterates t=0..511
// with a device-wide barrier between steps. Global A (h-state) loads
// are rotated one K-block ahead of the WMMA group; x_proj addends and
// c are preloaded before the K loop so their latency hides under WMMA.
// ---------------------------------------------------------------
__global__ __launch_bounds__(32)
void lstm_scan(const float*    __restrict__ xproj,   // [T, B, 4H]
               const _Float16* __restrict__ whh_t,   // [2048, 512]
               _Float16*       __restrict__ hA,      // [B, H] f16 ping
               _Float16*       __restrict__ hB,      // [B, H] f16 pong
               float*          __restrict__ c,       // [B, H] f32 (in place)
               float*          __restrict__ out,     // d_out
               unsigned*       __restrict__ bar) {
  extern __shared__ _Float16 smem[];   // 4 gates * 16 kblk * 32 lanes * 16 halves = 64KB
  const int lane = threadIdx.x;        // 0..31 (one wave)
  const int m0 = blockIdx.x * 16;      // batch tile
  const int n0 = blockIdx.y * 16;      // hidden tile
  const int row = lane & 15;
  const int hi  = lane >> 4;

  // ---- stage time-invariant B tiles into LDS (once) ----
#pragma unroll
  for (int g = 0; g < 4; ++g) {
    for (int kk = 0; kk < 16; ++kk) {
      const _Float16* p = whh_t + (size_t)(g * HH + n0 + row) * HH + kk * 32 + hi * 8;
      _Float16* dst = smem + ((size_t)(g * 16 + kk) * 32 + lane) * 16;
      *reinterpret_cast<v8h*>(dst)     = *reinterpret_cast<const v8h*>(p);
      *reinterpret_cast<v8h*>(dst + 8) = *reinterpret_cast<const v8h*>(p + 16);
    }
  }
  __syncthreads();

  const int rlo = hi * 8;
  const int cl  = lane & 15;
  const size_t HSEQ = (size_t)BB * TT * HH;

  for (int t = 0; t < TT; ++t) {
    const _Float16* h_in  = (t & 1) ? hB : hA;
    _Float16*       h_out = (t & 1) ? hA : hB;

    // ---- preload per-step elementwise operands (latency hidden by WMMAs) ----
    const float* xp = xproj + (size_t)t * (BB * GG);
    float xf[8], xi[8], xo[8], xg[8], cold[8];
#pragma unroll
    for (int j = 0; j < 8; ++j) {
      int b   = m0 + j + rlo;
      int col = n0 + cl;
      size_t gbase = (size_t)b * GG;
      xf[j] = xp[gbase + 0 * HH + col];
      xi[j] = xp[gbase + 1 * HH + col];
      xo[j] = xp[gbase + 2 * HH + col];
      xg[j] = xp[gbase + 3 * HH + col];
      cold[j] = c[(size_t)b * HH + col];
    }

    v8f acc[4];
#pragma unroll
    for (int g = 0; g < 4; ++g) acc[g] = (v8f)(0.0f);

    const _Float16* abase = h_in + (size_t)m0 * HH;
    v16h a_cur = load_tile_16x32(abase, HH, lane);
#pragma unroll
    for (int kk = 0; kk < 16; ++kk) {
      v16h a_nxt;
      if (kk < 15)
        a_nxt = load_tile_16x32(abase + (kk + 1) * 32, HH, lane);
      // B tiles from LDS (DScnt path, decoupled from global LOADcnt)
      v16h bt[4];
#pragma unroll
      for (int g = 0; g < 4; ++g) {
        const _Float16* bp = smem + ((size_t)(g * 16 + kk) * 32 + lane) * 16;
        bt[g] = pack16(*reinterpret_cast<const v8h*>(bp),
                       *reinterpret_cast<const v8h*>(bp + 8));
      }
#pragma unroll
      for (int g = 0; g < 4; ++g)
        acc[g] = __builtin_amdgcn_wmma_f32_16x16x32_f16(
            false, a_cur, false, bt[g], (short)0, acc[g], false, false);
      if (kk < 15) a_cur = a_nxt;
    }

    // ---- fused LSTM cell epilogue ----
#pragma unroll
    for (int j = 0; j < 8; ++j) {
      int b   = m0 + j + rlo;
      int col = n0 + cl;
      float fx = acc[0][j] + xf[j];
      float ix = acc[1][j] + xi[j];
      float ox = acc[2][j] + xo[j];
      float gx = acc[3][j] + xg[j];

      size_t ci = (size_t)b * HH + col;
      float cNew = sigmoidf_dev(fx) * cold[j] + sigmoidf_dev(ix) * tanhf(gx);
      float hNew = sigmoidf_dev(ox) * tanhf(cNew);

      c[ci] = cNew;
      h_out[ci] = (_Float16)hNew;
      out[(size_t)b * (TT * HH) + (size_t)t * HH + col] = hNew;   // h_seq
      if (t == TT - 1) {
        out[HSEQ + ci] = hNew;                   // h_T
        out[HSEQ + (size_t)BB * HH + ci] = cNew; // c_T
      }
    }

    // ---- device-wide barrier between steps (skip after last) ----
    if (t != TT - 1) {
      __builtin_amdgcn_fence(__ATOMIC_RELEASE, "agent");   // flush h_out stores
      if (lane == 0) {
        __hip_atomic_fetch_add(bar, 1u, __ATOMIC_RELAXED, __HIP_MEMORY_SCOPE_AGENT);
        const unsigned target = (unsigned)(t + 1) * (unsigned)NBLK_SCAN;
        while (__hip_atomic_load(bar, __ATOMIC_RELAXED, __HIP_MEMORY_SCOPE_AGENT) < target)
          __builtin_amdgcn_s_sleep(1);
      }
      __builtin_amdgcn_fence(__ATOMIC_ACQUIRE, "agent");   // invalidate stale h lines
    }
  }
}

// ---------------- host-side launch ----------------
extern "C" void kernel_launch(void* const* d_in, const int* in_sizes, int n_in,
                              void* d_out, int out_size, void* d_ws, size_t ws_size,
                              hipStream_t stream) {
  const float* x    = (const float*)d_in[0];   // [B, T, D]
  const float* wih  = (const float*)d_in[1];   // [D, 4H]
  const float* whh  = (const float*)d_in[2];   // [H, 4H]
  const float* bias = (const float*)d_in[3];   // [4H]
  float* out = (float*)d_out;

  // Workspace carve-up (bytes)
  char* ws = (char*)d_ws;
  const size_t XPROJ_BYTES = (size_t)TT * BB * GG * sizeof(float);      // 256 MB
  const size_t X16_BYTES   = (size_t)MROWS * DD * sizeof(_Float16);     // 32 MB
  const size_t WT_BYTES    = (size_t)GG * DD * sizeof(_Float16);        // 2 MB
  const size_t H16_BYTES   = (size_t)BB * HH * sizeof(_Float16);
  const size_t C_BYTES     = (size_t)BB * HH * sizeof(float);

  float*    xproj = (float*)ws;                         ws += XPROJ_BYTES;
  _Float16* x16   = (_Float16*)ws;                      ws += X16_BYTES;
  _Float16* wihT  = (_Float16*)ws;                      ws += WT_BYTES;
  _Float16* whhT  = (_Float16*)ws;                      ws += WT_BYTES;
  _Float16* hA    = (_Float16*)ws;                      ws += H16_BYTES;
  _Float16* hB    = (_Float16*)ws;                      ws += H16_BYTES;
  float*    cbuf  = (float*)ws;                         ws += C_BYTES;
  unsigned* bar   = (unsigned*)ws;                      ws += 256;
  (void)ws_size; (void)in_sizes; (void)n_in; (void)out_size;

  // 1) convert x to f16
  {
    int n = MROWS * DD;
    cvt_f32_to_f16<<<dim3((n + 255) / 256), dim3(256), 0, stream>>>(x, x16, n);
  }
  // 2) transpose + convert both weight matrices: [512,2048] -> [2048,512] f16
  {
    int n = DD * GG;
    transpose_w_f16<<<dim3((n + 255) / 256), dim3(256), 0, stream>>>(wih, wihT, DD, GG);
    transpose_w_f16<<<dim3((n + 255) / 256), dim3(256), 0, stream>>>(whh, whhT, HH, GG);
  }
  // 3) zero h0 / c0 state + barrier counter
  {
    int n = BB * HH;
    init_state<<<dim3((n + 255) / 256), dim3(256), 0, stream>>>(hA, hB, cbuf, bar, n);
  }
  // 4) parallel input projection GEMM (WMMA, 2x4 register-blocked)
  gemm_xproj<<<dim3(MROWS / 32, 8), dim3(128), 0, stream>>>(x16, wihT, bias, xproj);

  // 5) persistent cooperative scan: 128 single-wave blocks, 64KB dynamic LDS each
  lstm_scan<<<dim3(BB / 16, HH / 16), dim3(32), 65536, stream>>>(
      xproj, whhT, hA, hB, cbuf, out, bar);
}
